// SSM_27582279975483
// MI455X (gfx1250) — matrix operand order repaired and checked by reading, not compiled, not run on hardware
//
#include <hip/hip_runtime.h>
#include <hip/hip_bf16.h>

#ifndef __has_builtin
#define __has_builtin(x) 0
#endif

#if __has_builtin(__builtin_amdgcn_global_load_async_to_lds_b128)
#define USE_ASYNC_LDS 1
#endif

typedef __attribute__((ext_vector_type(16))) _Float16 v16h;
typedef __attribute__((ext_vector_type(8)))  _Float16 v8h;
typedef __attribute__((ext_vector_type(8)))  float    v8f;
typedef __attribute__((ext_vector_type(4)))  float    v4f;
typedef int v4i __attribute__((vector_size(16)));   // matches builtin param type

namespace {
constexpr int   kBa   = 256;    // batch
constexpr int   kT    = 2048;   // time steps
constexpr int   kI    = 128;    // input dim
constexpr int   kH    = 128;    // hidden dim
constexpr int   kN2   = 256;    // 2*H columns
constexpr int   kMT   = 16;     // batch rows per workgroup
constexpr int   kAROW = 264;    // padded A-buffer row (f16 elems), 528B = 33*16B
constexpr int   kPROW = 260;    // padded pre-buffer row (f32 elems)
constexpr float kGmin = 0.05f;
constexpr float kGmax = 0.95f;
}

// ---- async global->LDS copy (CDNA5 ASYNCcnt path), with safe fallback ----
__device__ __forceinline__ void cp16(const float* g, float* l) {
#ifdef USE_ASYNC_LDS
  __builtin_amdgcn_global_load_async_to_lds_b128((v4i*)g, (v4i*)l, 0, 0);
#else
  *(v4f*)l = *(const v4f*)g;
#endif
}

__device__ __forceinline__ void cp_wait() {
#ifdef USE_ASYNC_LDS
#if __has_builtin(__builtin_amdgcn_s_wait_asynccnt)
  __builtin_amdgcn_s_wait_asynccnt(0);
#else
  asm volatile("s_wait_asynccnt 0" ::: "memory");
#endif
#endif
}

// Branch-free tanh: copysign(1 - 2/(e^{2|x|}+1), x). Saturates via inf, no
// EXEC-mask games in the scan's critical path (ocml tanhf is branchy).
__device__ __forceinline__ float fast_tanh(float x) {
  const float e = __expf(2.f * fabsf(x));
  const float t = 1.f - 2.f / (e + 1.f);
  return copysignf(t, x);
}

// Persistent fused SSM scan. Grid = 16 workgroups (one per 16 batch rows),
// 256 threads = 8 wave32. Each wave owns N-columns [w*32, w*32+32) and keeps
// its W slice (f16, full K=256) in 128 VGPRs for the whole scan.
__global__ __launch_bounds__(256, 1)
void ssm_scan_wmma(const float* __restrict__ x,
                   const float* __restrict__ fw,
                   const float* __restrict__ W,
                   const float* __restrict__ b,
                   float* __restrict__ out) {
  __shared__ _Float16 Abuf[kMT][kAROW];        // A rows: [x_t*e^fw (K 0..127) | h (K 128..255)]
  __shared__ float    preBuf[kMT][kPROW];      // per-step matmul result (16 x 256)
  __shared__ float    xstage[2][kMT][kI];      // double-buffered async x staging
  __shared__ float    biasS[kN2];
  __shared__ float    expwS[kI];

  const int tid  = threadIdx.x;
  const int lane = tid & 31;
  const int wv   = tid >> 5;                   // wave id 0..7
  const int m0   = blockIdx.x * kMT;           // batch row base

  // ---------------- init ----------------
  if (tid < kN2) biasS[tid] = b[tid];
  if (tid < kI)  expwS[tid] = __expf(fw[tid]);
  for (int q = tid; q < kMT * kH; q += 256) {  // h0 = 0 in A-buffer h-half
    Abuf[q >> 7][kI + (q & 127)] = (_Float16)0.f;
  }

  // Preload this wave's B fragments: W[k][n] f32 -> f16 registers.
  // Fragment (kk, ntile): 32 K-rows x 16 N-cols; lane holds col lane&15, K rows k0+16*(lane>>4)+0..15.
  const int nb0 = wv * 32;
  v16h Bf0[8], Bf1[8];
#pragma unroll
  for (int kk = 0; kk < 8; ++kk) {
    const float* p0 = W + (size_t)(kk * 32 + ((lane >> 4) << 4)) * kN2 + nb0 + (lane & 15);
    v16h t0, t1;
#pragma unroll
    for (int i = 0; i < 16; ++i) {
      t0[i] = (_Float16)p0[(size_t)i * kN2];
      t1[i] = (_Float16)p0[(size_t)i * kN2 + 16];
    }
    Bf0[kk] = t0;
    Bf1[kk] = t1;
  }

  // Per-thread hidden-state registers: thread owns row um, cols uc0+16j.
  const int um  = tid >> 4;
  const int uc0 = tid & 15;
  float hreg[8];
#pragma unroll
  for (int j = 0; j < 8; ++j) hreg[j] = 0.f;

  // Prefetch x(t=0) into xstage[0]: 512 float4 tiles, 2 per thread.
#pragma unroll
  for (int k = 0; k < 2; ++k) {
    const int q = tid + k * 256;
    const int m = q >> 5, i = (q & 31) * 4;
    cp16(x + ((size_t)(m0 + m) * kT + 0) * kI + i, &xstage[0][m][i]);
  }
  cp_wait();
  __syncthreads();

  // ---------------- scan ----------------
  for (int t = 0; t < kT; ++t) {
    const int cur = t & 1, nxt = cur ^ 1;

    // Stage x_t: scale by e^fw, convert to f16 into A-buffer low half.
#pragma unroll
    for (int k = 0; k < 2; ++k) {
      const int q = tid + k * 256;
      const int m = q >> 5, i = (q & 31) * 4;
      v4f v = *(const v4f*)&xstage[cur][m][i];
      Abuf[m][i + 0] = (_Float16)(v.x * expwS[i + 0]);
      Abuf[m][i + 1] = (_Float16)(v.y * expwS[i + 1]);
      Abuf[m][i + 2] = (_Float16)(v.z * expwS[i + 2]);
      Abuf[m][i + 3] = (_Float16)(v.w * expwS[i + 3]);
    }

    // Async prefetch x(t+1) while this step's WMMAs run.
    if (t + 1 < kT) {
#pragma unroll
      for (int k = 0; k < 2; ++k) {
        const int q = tid + k * 256;
        const int m = q >> 5, i = (q & 31) * 4;
        cp16(x + ((size_t)(m0 + m) * kT + (t + 1)) * kI + i, &xstage[nxt][m][i]);
      }
    }
    __syncthreads();  // A-buffer (x half + h half from prev update) ready

    // pre(16x32 cols of this wave) = A(16x256) @ W(256x32), K in 8 steps of 32.
    v8f acc0 = {};
    v8f acc1 = {};
    const _Float16* ap = &Abuf[lane & 15][(lane >> 4) << 4];
#pragma unroll
    for (int kk = 0; kk < 8; ++kk) {
      v8h lo = *(const v8h*)(ap + kk * 32);
      v8h hi = *(const v8h*)(ap + kk * 32 + 8);
      v16h a = __builtin_shufflevector(lo, hi, 0, 1, 2, 3, 4, 5, 6, 7,
                                       8, 9, 10, 11, 12, 13, 14, 15);
      acc0 = __builtin_amdgcn_wmma_f32_16x16x32_f16(false, a, false, Bf0[kk],
                                                    (short)0, acc0, false, false);
      acc1 = __builtin_amdgcn_wmma_f32_16x16x32_f16(false, a, false, Bf1[kk],
                                                    (short)0, acc1, false, false);
    }

    // Scatter accumulators to preBuf (documented C layout: lane<16 -> M=r,
    // lane>=16 -> M=8+r; N = nb0 + (lane&15)).
    {
      const int mb = (lane < 16) ? 0 : 8;
      const int nn = lane & 15;
#pragma unroll
      for (int r = 0; r < 8; ++r) {
        preBuf[mb + r][nb0 + nn]      = acc0[r];
        preBuf[mb + r][nb0 + 16 + nn] = acc1[r];
      }
    }
    __syncthreads();  // preBuf complete across waves

    // Elementwise: g = clip(sigmoid(.)), d = tanh(.), h = (1-g)h + g d.
    {
      float* orow = out + ((size_t)(m0 + um) * kT + t) * kH;
#pragma unroll
      for (int j = 0; j < 8; ++j) {
        const int c = uc0 + j * 16;
        const float pg = preBuf[um][c] + biasS[c];
        const float pd = preBuf[um][kH + c] + biasS[kH + c];
        float g = 1.f / (1.f + __expf(-pg));
        g = fminf(fmaxf(g, kGmin), kGmax);
        const float d = fast_tanh(pd);
        const float hn = (1.f - g) * hreg[j] + g * d;
        hreg[j] = hn;
        Abuf[um][kI + c] = (_Float16)hn;  // feed h into next step's A
        orow[c] = hn;                     // outputs[b][t][c]
      }
    }
    cp_wait();        // next x tile landed in LDS (this wave's portion)
    __syncthreads();  // h-half of A-buffer + preBuf reuse safe
  }

  // Final hidden state h_T appended after outputs.
  {
    float* hrow = out + (size_t)kBa * kT * kH + (size_t)(m0 + um) * kH;
#pragma unroll
    for (int j = 0; j < 8; ++j) hrow[uc0 + j * 16] = hreg[j];
  }
}

extern "C" void kernel_launch(void* const* d_in, const int* in_sizes, int n_in,
                              void* d_out, int out_size, void* d_ws, size_t ws_size,
                              hipStream_t stream) {
  (void)in_sizes; (void)n_in; (void)out_size; (void)d_ws; (void)ws_size;
  const float* x  = (const float*)d_in[0];
  const float* fw = (const float*)d_in[1];
  const float* W  = (const float*)d_in[2];
  const float* b  = (const float*)d_in[3];
  float* out = (float*)d_out;

  dim3 grid(kBa / kMT);   // 16 workgroups, one per 16 batch rows
  dim3 block(256);        // 8 wave32
  ssm_scan_wmma<<<grid, block, 0, stream>>>(x, fw, W, b, out);
}
